// RoPEMultiHeadAttention_24369644437756
// MI455X (gfx1250) — matrix operand order repaired and checked by reading, not compiled, not run on hardware
//
#include <hip/hip_runtime.h>

// ---------------- types ----------------
typedef __attribute__((ext_vector_type(16))) _Float16 v16h;
typedef __attribute__((ext_vector_type(8)))  _Float16 v8h;
typedef __attribute__((ext_vector_type(8)))  float    v8f;

#define D_MODEL 1024
#define N_HEADS 16
#define HEAD_DIM 64
#define SEQ_T 2048
#define BATCH 2
#define NTOK (BATCH * SEQ_T)   // 4096

__device__ __forceinline__ v8f wmma_f16(v16h a, v16h b, v8f c) {
  // D = A(16x32) * B(32x16) + C, f32 accumulate
  return __builtin_amdgcn_wmma_f32_16x16x32_f16(
      /*neg_a=*/false, a, /*neg_b=*/false, b,
      /*c_mod=*/(short)0, c, /*reuse_a=*/false, /*reuse_b=*/false);
}

// A-fragment loader: lane holds K {2*kb..2*kb+15} with kb = (lane>>4)*8
template <typename P>
__device__ __forceinline__ v16h load_afrag(const P* p, int kb) {
  v8h lo = *(const v8h*)(p + 2 * kb);
  v8h hi = *(const v8h*)(p + 2 * kb + 8);
  v16h a;
#pragma unroll
  for (int i = 0; i < 8; ++i) { a[i] = lo[i]; a[8 + i] = hi[i]; }
  return a;
}

// CDNA5 async global->LDS copy, 16 bytes per lane (GLOBAL_LOAD_ASYNC_TO_LDS_B128).
// Tracked by ASYNCcnt; wait with s_wait_asynccnt.
__device__ __forceinline__ void async_cp16(unsigned lds_off, const _Float16* gsrc) {
  asm volatile("global_load_async_to_lds_b128 %0, %1, off"
               :: "v"(lds_off), "v"((unsigned long long)(size_t)gsrc)
               : "memory");
}
__device__ __forceinline__ void async_wait0() {
  asm volatile("s_wait_asynccnt 0x0" ::: "memory");
}

// ---------------- fp32 -> fp16 convert ----------------
__global__ void cvt_f32_f16(const float* __restrict__ s, _Float16* __restrict__ d, int n) {
  int i = blockIdx.x * blockDim.x + threadIdx.x;
  if (i < n) d[i] = (_Float16)s[i];
}

// ---------------- fused GEMM (+bias, +RoPE, layout stores) ----------------
// A: [4096,1024] fp16 row-major.  W: [1024,1024] fp16 row-major (used as W^T).
// One wave computes a 32(M) x 64(N) tile (B fragments reused across 2 A fragments).
// mode 0/1: RoPE, store fp16 [B,H,T,64]   (Q / K)
// mode 2:   store fp16 transposed [B,H,64,T]  (V^T)
// mode 3:   store fp32 [4096,1024]        (final out)
__global__ __launch_bounds__(32)
void gemm_wmma(const _Float16* __restrict__ A, const _Float16* __restrict__ W,
               const float* __restrict__ bias, void* __restrict__ outp, int mode) {
  const int lane = threadIdx.x;
  const int nl = lane & 15;       // N position in frag / C-layout column
  const int g  = lane >> 4;       // half-wave id
  const int kb = g * 8;
  const int mt = blockIdx.x;      // 0..127 (32 rows each)
  const int nt = blockIdx.y;      // 0..15  (== head index for modes 0..2)

  v8f acc[2][4] = {};
  const _Float16* Arow0 = A + (size_t)(mt * 32 + nl) * D_MODEL;
  const _Float16* Arow1 = Arow0 + (size_t)16 * D_MODEL;

  for (int k0 = 0; k0 < D_MODEL; k0 += 32) {
    v16h a0 = load_afrag(Arow0 + k0, kb);
    v16h a1 = load_afrag(Arow1 + k0, kb);
#pragma unroll
    for (int f = 0; f < 4; ++f) {
      // B[k][n] = W[n][k]: lane nl holds 16 consecutive k of row (nt*64+f*16+nl)
      const v16h b = *(const v16h*)(W + (size_t)(nt * 64 + f * 16 + nl) * D_MODEL + k0 + g * 16);
      acc[0][f] = wmma_f16(a0, b, acc[0][f]);
      acc[1][f] = wmma_f16(a1, b, acc[1][f]);
    }
  }

  const float nlog = -logf(10000.0f) / 32.0f;
#pragma unroll
  for (int mi = 0; mi < 2; ++mi) {
    const int gmBase = mt * 32 + mi * 16 + 8 * g;  // + r gives global token row

    // bias
#pragma unroll
    for (int f = 0; f < 4; ++f) {
      float bv = bias[nt * 64 + f * 16 + nl];
#pragma unroll
      for (int r = 0; r < 8; ++r) acc[mi][f][r] += bv;
    }

    if (mode <= 1) {
      // RoPE in C-layout: partner d <-> d+32 lives in fragment f <-> f+2, same lane/reg.
#pragma unroll
      for (int fp = 0; fp < 2; ++fp) {
        float invf = __expf(nlog * (float)(fp * 16 + nl));  // inv_freq[d % 32]
#pragma unroll
        for (int r = 0; r < 8; ++r) {
          int t = (gmBase + r) & (SEQ_T - 1);
          float ang = (float)t * invf;
          float c = __cosf(ang), s = __sinf(ang);
          float x1 = acc[mi][fp][r], x2 = acc[mi][fp + 2][r];
          acc[mi][fp][r]     = x1 * c - x2 * s;
          acc[mi][fp + 2][r] = x2 * c + x1 * s;
        }
      }
      // fp16 store [B,H,T,64]; h == nt, d == f*16+nl
      _Float16* o = (_Float16*)outp;
#pragma unroll
      for (int f = 0; f < 4; ++f) {
        int dd = f * 16 + nl;
#pragma unroll
        for (int r = 0; r < 8; ++r) {
          int gm = gmBase + r;
          int bi = gm >> 11, t = gm & (SEQ_T - 1);
          o[(((size_t)bi * N_HEADS + nt) * SEQ_T + t) * HEAD_DIM + dd] = (_Float16)acc[mi][f][r];
        }
      }
    } else if (mode == 2) {
      // fp16 store V^T [B,H,64,T]
      _Float16* o = (_Float16*)outp;
#pragma unroll
      for (int f = 0; f < 4; ++f) {
        int dd = f * 16 + nl;
#pragma unroll
        for (int r = 0; r < 8; ++r) {
          int gm = gmBase + r;
          int bi = gm >> 11, t = gm & (SEQ_T - 1);
          o[(((size_t)bi * N_HEADS + nt) * HEAD_DIM + dd) * SEQ_T + t] = (_Float16)acc[mi][f][r];
        }
      }
    } else {
      // fp32 store [4096,1024]
      float* o = (float*)outp;
#pragma unroll
      for (int f = 0; f < 4; ++f) {
#pragma unroll
        for (int r = 0; r < 8; ++r) {
          o[(size_t)(gmBase + r) * D_MODEL + nt * 64 + f * 16 + nl] = acc[mi][f][r];
        }
      }
    }
  }
}

// ---------------- flash attention ----------------
// 128-thread block = 4 waves; each wave owns 32 queries (2 tiles). The block
// cooperatively stages each 32-key K chunk and V^T chunk into DOUBLE-BUFFERED
// LDS via async copies: chunk i+1 fills while chunk i computes, with one
// s_wait_asynccnt + one barrier per iteration.
// Q,K: fp16 [B,H,T,64] (RoPE applied). Vt: fp16 [B,H,64,T]. O: fp16 [B,T,H*64].
#define LKP 72  // ldsK row pitch (halfs): 144B stride -> conflict-free 16-lane reads
#define LVP 40  // ldsV row pitch (halfs): 80B stride
#define LPP 40  // ldsP row pitch (halfs)

__global__ __launch_bounds__(128)
void attn_wmma(const _Float16* __restrict__ Q, const _Float16* __restrict__ K,
               const _Float16* __restrict__ Vt, _Float16* __restrict__ O) {
  const int tid = threadIdx.x;
  const int lane = tid & 31, wid = tid >> 5;
  const int nl = lane & 15, g = lane >> 4, kb = g * 8;
  const int qb = blockIdx.x, h = blockIdx.y, b = blockIdx.z;

  const _Float16* Qh = Q  + ((size_t)b * N_HEADS + h) * SEQ_T * HEAD_DIM;
  const _Float16* Kh = K  + ((size_t)b * N_HEADS + h) * SEQ_T * HEAD_DIM;
  const _Float16* Vh = Vt + ((size_t)b * N_HEADS + h) * HEAD_DIM * SEQ_T;

  __shared__ __align__(16) _Float16 ldsK[2][32 * LKP];
  __shared__ __align__(16) _Float16 ldsV[2][64 * LVP];
  __shared__ __align__(16) _Float16 ldsP[4 * 16 * LPP];
  _Float16* myP = &ldsP[wid * 16 * LPP];

  const int q0 = qb * 128 + wid * 32;

  // Q fragments, pre-scaled by 1/sqrt(64) = 0.125 (exact in fp16)
  v16h aq[2][2];
#pragma unroll
  for (int ti = 0; ti < 2; ++ti) {
    const _Float16* qp = Qh + (size_t)(q0 + ti * 16 + nl) * HEAD_DIM;
    aq[ti][0] = load_afrag(qp, kb);
    aq[ti][1] = load_afrag(qp + 32, kb);
#pragma unroll
    for (int i = 0; i < 16; ++i) {
      aq[ti][0][i] = aq[ti][0][i] * (_Float16)0.125f;
      aq[ti][1][i] = aq[ti][1][i] * (_Float16)0.125f;
    }
  }

  v8f o[2][4] = {};
  float m[2][8], l[2][8];
#pragma unroll
  for (int ti = 0; ti < 2; ++ti)
#pragma unroll
    for (int r = 0; r < 8; ++r) { m[ti][r] = -1e30f; l[ti][r] = 0.0f; }

  // cooperative async stage of one 32-key chunk into buffer bufi
  auto stage = [&](int bufi, int kc) {
#pragma unroll
    for (int c = tid; c < 256; c += 128) {
      int kr = c >> 3, ks = c & 7;   // K: 32 rows x 8 segs of 8 halfs
      async_cp16((unsigned)(size_t)&ldsK[bufi][kr * LKP + ks * 8],
                 Kh + (size_t)(kc + kr) * HEAD_DIM + ks * 8);
      int vr = c >> 2, vs = c & 3;   // V: 64 rows x 4 segs of 8 halfs
      async_cp16((unsigned)(size_t)&ldsV[bufi][vr * LVP + vs * 8],
                 Vh + (size_t)vr * SEQ_T + kc + vs * 8);
    }
  };

  // prologue: fill buffer 0
  stage(0, 0);
  async_wait0();
  __syncthreads();

  int cur = 0;
  for (int kc = 0; kc < SEQ_T; kc += 32) {
    // overlap: fill the other buffer with the next chunk while computing
    if (kc + 32 < SEQ_T) stage(cur ^ 1, kc + 32);

    // hoist K/V fragments for this chunk (shared by both query tiles)
    v16h kf[2][2], vf[4];
#pragma unroll
    for (int j = 0; j < 2; ++j) {
      const _Float16* kp = &ldsK[cur][(j * 16 + nl) * LKP + g * 16];
      kf[j][0] = *(const v16h*)(kp);
      kf[j][1] = *(const v16h*)(kp + 32);
    }
#pragma unroll
    for (int f = 0; f < 4; ++f)
      vf[f] = *(const v16h*)&ldsV[cur][(f * 16 + nl) * LVP + g * 16];

#pragma unroll
    for (int ti = 0; ti < 2; ++ti) {
      // ---- scores: two 16x16 tiles over this 32-key chunk ----
      v8f s0 = {}, s1 = {};
      s0 = wmma_f16(aq[ti][0], kf[0][0], s0);
      s0 = wmma_f16(aq[ti][1], kf[0][1], s0);
      s1 = wmma_f16(aq[ti][0], kf[1][0], s1);
      s1 = wmma_f16(aq[ti][1], kf[1][1], s1);

      // ---- online softmax in C-layout (row r+8g lives in reg r) ----
      float al[8], p0[8], p1[8];
#pragma unroll
      for (int r = 0; r < 8; ++r) {
        float t = fmaxf(s0[r], s1[r]);
        t = fmaxf(t, __shfl_xor(t, 1, 32));
        t = fmaxf(t, __shfl_xor(t, 2, 32));
        t = fmaxf(t, __shfl_xor(t, 4, 32));
        t = fmaxf(t, __shfl_xor(t, 8, 32));
        float mn = fmaxf(m[ti][r], t);
        al[r] = __expf(m[ti][r] - mn);
        m[ti][r] = mn;
        p0[r] = __expf(s0[r] - mn);
        p1[r] = __expf(s1[r] - mn);
        float u = p0[r] + p1[r];
        u += __shfl_xor(u, 1, 32);
        u += __shfl_xor(u, 2, 32);
        u += __shfl_xor(u, 4, 32);
        u += __shfl_xor(u, 8, 32);
        l[ti][r] = l[ti][r] * al[r] + u;
      }
#pragma unroll
      for (int f = 0; f < 4; ++f)
#pragma unroll
        for (int r = 0; r < 8; ++r) o[ti][f][r] *= al[r];

      // ---- C-layout -> A-fragment re-layout via per-wave LDS tile ----
#pragma unroll
      for (int r = 0; r < 8; ++r) {
        myP[(r + 8 * g) * LPP + nl]      = (_Float16)p0[r];
        myP[(r + 8 * g) * LPP + 16 + nl] = (_Float16)p1[r];
      }
      v16h pa = load_afrag(&myP[nl * LPP], kb);

      // ---- O += P * V ----
#pragma unroll
      for (int f = 0; f < 4; ++f) o[ti][f] = wmma_f16(pa, vf[f], o[ti][f]);
    }

    async_wait0();     // my fills of the next buffer have landed in LDS
    __syncthreads();   // everyone done reading cur & done staging next
    cur ^= 1;
  }

  // ---- finalize + store fp16 [B,T,H*64] ----
#pragma unroll
  for (int ti = 0; ti < 2; ++ti) {
    float inv[8];
#pragma unroll
    for (int r = 0; r < 8; ++r) inv[r] = 1.0f / l[ti][r];
#pragma unroll
    for (int f = 0; f < 4; ++f) {
      int dd = h * 64 + f * 16 + nl;
#pragma unroll
      for (int r = 0; r < 8; ++r) {
        int t = q0 + ti * 16 + 8 * g + r;
        O[((size_t)b * SEQ_T + t) * D_MODEL + dd] = (_Float16)(o[ti][f][r] * inv[r]);
      }
    }
  }
}

// ---------------- host launch ----------------
extern "C" void kernel_launch(void* const* d_in, const int* in_sizes, int n_in,
                              void* d_out, int out_size, void* d_ws, size_t ws_size,
                              hipStream_t stream) {
  (void)in_sizes; (void)n_in; (void)out_size; (void)ws_size;
  const float* x  = (const float*)d_in[0];
  const float* Wq = (const float*)d_in[1];
  const float* bq = (const float*)d_in[2];
  const float* Wk = (const float*)d_in[3];
  const float* bk = (const float*)d_in[4];
  const float* Wv = (const float*)d_in[5];
  const float* bv = (const float*)d_in[6];
  const float* Wo = (const float*)d_in[7];
  const float* bo = (const float*)d_in[8];

  char* w = (char*)d_ws;
  size_t off = 0;
  auto take = [&](size_t nelem) -> _Float16* {
    _Float16* p = (_Float16*)(w + off);
    off += (nelem * sizeof(_Float16) + 255) & ~(size_t)255;
    return p;
  };
  _Float16* xh  = take((size_t)NTOK * D_MODEL);      // x fp16
  _Float16* wqh = take((size_t)D_MODEL * D_MODEL);
  _Float16* wkh = take((size_t)D_MODEL * D_MODEL);
  _Float16* wvh = take((size_t)D_MODEL * D_MODEL);
  _Float16* woh = take((size_t)D_MODEL * D_MODEL);
  _Float16* qh  = take((size_t)NTOK * D_MODEL);      // [B,H,T,64]
  _Float16* kh  = take((size_t)NTOK * D_MODEL);      // [B,H,T,64]
  _Float16* vth = take((size_t)NTOK * D_MODEL);      // [B,H,64,T]
  _Float16* ath = take((size_t)NTOK * D_MODEL);      // attn out [B,T,1024]

  const int NX = NTOK * D_MODEL;       // 4194304
  const int NW = D_MODEL * D_MODEL;    // 1048576
  cvt_f32_f16<<<(NX + 255) / 256, 256, 0, stream>>>(x,  xh,  NX);
  cvt_f32_f16<<<(NW + 255) / 256, 256, 0, stream>>>(Wq, wqh, NW);
  cvt_f32_f16<<<(NW + 255) / 256, 256, 0, stream>>>(Wk, wkh, NW);
  cvt_f32_f16<<<(NW + 255) / 256, 256, 0, stream>>>(Wv, wvh, NW);
  cvt_f32_f16<<<(NW + 255) / 256, 256, 0, stream>>>(Wo, woh, NW);

  dim3 gg(NTOK / 32, D_MODEL / 64, 1);  // (128,16)
  gemm_wmma<<<gg, 32, 0, stream>>>(xh, wqh, bq, (void*)qh,  0);  // Q + RoPE
  gemm_wmma<<<gg, 32, 0, stream>>>(xh, wkh, bk, (void*)kh,  1);  // K + RoPE
  gemm_wmma<<<gg, 32, 0, stream>>>(xh, wvh, bv, (void*)vth, 2);  // V^T

  attn_wmma<<<dim3(SEQ_T / 128, N_HEADS, BATCH), 128, 0, stream>>>(qh, kh, vth, ath);

  gemm_wmma<<<gg, 32, 0, stream>>>(ath, woh, bo, d_out, 3);      // out proj, fp32
}
